// VLMUpdateModule_25426206392333
// MI455X (gfx1250) — matrix operand order repaired and checked by reading, not compiled, not run on hardware
//
#include <hip/hip_runtime.h>

typedef __attribute__((ext_vector_type(16))) __bf16 v16bf;
typedef __attribute__((ext_vector_type(8)))  float  v8f;

#define D_OBS 1536
#define D_VLM 2048
#define T_SEQ 512
#define PITCH 36      // LDS row pitch in bf16 units (72 bytes)
#define KSTEP 32
#define NSTEP (D_OBS / KSTEP)
#define EPS_F 1e-6f

#define BTSZ ((size_t)D_VLM * PITCH * 2)   // bytes per B panel buffer (147456)
#define ALSZ ((size_t)32 * PITCH * 2)      // bytes per A panel buffer (2304)

struct Frag32 { uint2 q[4]; };   // 32 bytes == v16bf

// single-instruction packed f32->bf16 (RNE); mnemonic verified in gfx1250 disasm
static __device__ __forceinline__ unsigned pkbf(float a, float b) {
    unsigned r;
    asm("v_cvt_pk_bf16_f32 %0, %1, %2" : "=v"(r) : "v"(a), "v"(b));
    return r;
}

// pack 8 rows x 4 cols of f32 into bf16, store transposed into BT[n][k]
static __device__ __forceinline__ void packStoreB(const float4* wv, __bf16* BTn,
                                                  int c0, int r0) {
    #pragma unroll
    for (int cc = 0; cc < 4; ++cc) {
        float e0 = ((const float*)&wv[0])[cc], e1 = ((const float*)&wv[1])[cc];
        float e2 = ((const float*)&wv[2])[cc], e3 = ((const float*)&wv[3])[cc];
        float e4 = ((const float*)&wv[4])[cc], e5 = ((const float*)&wv[5])[cc];
        float e6 = ((const float*)&wv[6])[cc], e7 = ((const float*)&wv[7])[cc];
        uint2 lo; lo.x = pkbf(e0, e1); lo.y = pkbf(e2, e3);
        uint2 hi; hi.x = pkbf(e4, e5); hi.y = pkbf(e6, e7);
        __bf16* dst = BTn + (size_t)(c0 + cc) * PITCH + r0;
        *(uint2*)(dst)     = lo;
        *(uint2*)(dst + 4) = hi;
    }
}

static __device__ __forceinline__ v16bf loadFragB(const __bf16* BTc, int n, int half) {
    const __bf16* bp = BTc + (size_t)n * PITCH + half * 8;
    Frag32 f;
    f.q[0] = *(const uint2*)(bp);
    f.q[1] = *(const uint2*)(bp + 4);
    f.q[2] = *(const uint2*)(bp + 16);
    f.q[3] = *(const uint2*)(bp + 20);
    return __builtin_bit_cast(v16bf, f);
}

__global__ __launch_bounds__(512, 1)
void vlm_fused_kernel(const float* __restrict__ past, const float* __restrict__ obs,
                      const int* __restrict__ cat,    const float* __restrict__ W,
                      const float* __restrict__ bias, const float* __restrict__ alog,
                      const float* __restrict__ trms, float* __restrict__ out)
{
    extern __shared__ char smem[];
    float* ST = (float*)(smem + 2 * BTSZ + 2 * ALSZ);   // [32][4] row stats

    const int tid   = threadIdx.x;
    const int lane  = tid & 31;
    const int w     = tid >> 5;            // wave 0..15
    const int g     = w >> 3;              // row-group 0..1 (16 t-rows each)
    const int nbase = (w & 7) * 256;       // this wave's N chunk
    const int half  = lane >> 4;           // h
    const int ln    = lane & 15;

    const int bb    = (int)blockIdx.x >> 4;
    const int t0    = ((int)blockIdx.x & 15) * 32;   // 32 t-rows per block
    const int c     = cat[bb];

    const float* Wc    = W    + (size_t)c  * D_OBS * D_VLM;
    const float* obsB  = obs  + (size_t)bb * T_SEQ * D_OBS;
    const float* pastB = past + (size_t)bb * T_SEQ * D_VLM;
    float*       outB  = out  + (size_t)bb * T_SEQ * D_VLM;

    if (tid < 128) ST[tid] = 0.0f;

    v8f zero = 0.0f;
    v8f acc[16];
    #pragma unroll
    for (int i = 0; i < 16; ++i) acc[i] = zero;

    const int c0  = tid * 4;          // 4 W columns owned by this thread (staging)
    const int rA  = tid >> 4;         // A staging row 0..31
    const int ksA = (tid & 15) * 2;   // A staging K pair

    // ---- prologue: stage panel 0 into buffer 0 ----
    {
        __bf16* BT0 = (__bf16*)smem;
        __bf16* AL0 = (__bf16*)(smem + 2 * BTSZ);
        #pragma unroll
        for (int q = 0; q < 4; ++q) {
            float4 wv[8];
            #pragma unroll
            for (int u = 0; u < 8; ++u)
                wv[u] = *(const float4*)(Wc + (size_t)(q * 8 + u) * D_VLM + c0);
            packStoreB(wv, BT0, c0, q * 8);
        }
        const float2 av = *(const float2*)(obsB + (size_t)(t0 + rA) * D_OBS + ksA);
        *(unsigned*)(AL0 + rA * PITCH + ksA) = pkbf(av.x, av.y);
    }
    __syncthreads();

    // ---- main pipelined K loop: compute panel `step`, stage panel `step+1` ----
    for (int step = 0; step < NSTEP; ++step) {
        const int  cur  = step & 1;
        const int  nxt  = cur ^ 1;
        const int  nkk  = (step + 1) * KSTEP;
        const bool more = (nkk < D_OBS);
        const __bf16* BTc = (const __bf16*)(smem + (size_t)cur * BTSZ);
        const __bf16* ALc = (const __bf16*)(smem + 2 * BTSZ + (size_t)cur * ALSZ);
        __bf16* BTn = (__bf16*)(smem + (size_t)nxt * BTSZ);
        __bf16* ALn = (__bf16*)(smem + 2 * BTSZ + (size_t)nxt * ALSZ);

        // A fragment for this wave's row group (ISA 16-bit 16x32 A layout)
        Frag32 fa;
        {
            const __bf16* ap = ALc + (size_t)(g * 16 + ln) * PITCH + half * 8;
            fa.q[0] = *(const uint2*)(ap);
            fa.q[1] = *(const uint2*)(ap + 4);
            fa.q[2] = *(const uint2*)(ap + 16);
            fa.q[3] = *(const uint2*)(ap + 20);
        }
        const v16bf a = __builtin_bit_cast(v16bf, fa);

        v16bf fb[2];
        fb[0] = loadFragB(BTc, nbase + ln, half);

        float2 avn;
        #pragma unroll
        for (int q = 0; q < 4; ++q) {
            float4 wv[8];
            if (more) {               // uniform branch: pipeline next panel's loads
                #pragma unroll
                for (int u = 0; u < 8; ++u)
                    wv[u] = *(const float4*)(Wc + (size_t)(nkk + q * 8 + u) * D_VLM + c0);
                if (q == 0)
                    avn = *(const float2*)(obsB + (size_t)(t0 + rA) * D_OBS + nkk + ksA);
            }
            #pragma unroll
            for (int t = 0; t < 4; ++t) {
                const int nt = q * 4 + t;
                if (nt < 15)          // preload next B fragment (double buffer)
                    fb[(nt + 1) & 1] = loadFragB(BTc, nbase + (nt + 1) * 16 + ln, half);
                acc[nt] = __builtin_amdgcn_wmma_f32_16x16x32_bf16(
                    false, a, false, fb[nt & 1], (short)0, acc[nt], false, false);
            }
            if (more) {
                packStoreB(wv, BTn, c0, q * 8);
                if (q == 0)
                    *(unsigned*)(ALn + rA * PITCH + ksA) = pkbf(avn.x, avn.y);
            }
        }
        __syncthreads();
    }

    // ---- fused epilogue: bias, row stats (So2, Sp2, Spo), closed-form rms blend ----
    const float alpha = 1.0f / (1.0f + __expf(-alog[c]));
    float tr = trms[c]; tr = fminf(fmaxf(tr, 4.0f), 200.0f);

    float so2[8], sp2[8], spo[8];
    #pragma unroll
    for (int v = 0; v < 8; ++v) { so2[v] = 0.f; sp2[v] = 0.f; spo[v] = 0.f; }

    #pragma unroll
    for (int nt = 0; nt < 16; ++nt) {
        const int n = nbase + nt * 16 + ln;
        const float bn = bias[(size_t)c * D_VLM + n];
        #pragma unroll
        for (int v = 0; v < 8; ++v) {
            const int t = t0 + g * 16 + v + 8 * half;
            float o = acc[nt][v] + bn;
            acc[nt][v] = o;
            float p = pastB[(size_t)t * D_VLM + n];
            so2[v] += o * o; sp2[v] += p * p; spo[v] += p * o;
        }
    }
    #pragma unroll
    for (int m = 8; m >= 1; m >>= 1) {
        #pragma unroll
        for (int v = 0; v < 8; ++v) {
            so2[v] += __shfl_xor(so2[v], m, 16);
            sp2[v] += __shfl_xor(sp2[v], m, 16);
            spo[v] += __shfl_xor(spo[v], m, 16);
        }
    }
    if (ln == 0) {
        #pragma unroll
        for (int v = 0; v < 8; ++v) {
            const int row = g * 16 + v + 8 * half;
            atomicAdd(&ST[row * 4 + 0], so2[v]);
            atomicAdd(&ST[row * 4 + 1], sp2[v]);
            atomicAdd(&ST[row * 4 + 2], spo[v]);
        }
    }
    __syncthreads();

    float Af[8], Bf[8], Sf[8];
    #pragma unroll
    for (int v = 0; v < 8; ++v) {
        const int row = g * 16 + v + 8 * half;
        const float S_o2 = ST[row * 4 + 0];
        const float S_p2 = ST[row * 4 + 1];
        const float S_po = ST[row * 4 + 2];
        const float inv_d = 1.0f / (float)D_VLM;
        const float rp = __fsqrt_rn(S_p2 * inv_d + EPS_F);
        const float ro = __fsqrt_rn(S_o2 * inv_d + EPS_F);
        const float Ac = alpha / rp;
        const float Bc = (1.0f - alpha) / ro;
        const float sx2 = Ac * Ac * S_p2 + Bc * Bc * S_o2 + 2.0f * Ac * Bc * S_po;
        const float rx = __fsqrt_rn(sx2 * inv_d + EPS_F);
        Af[v] = Ac; Bf[v] = Bc; Sf[v] = tr / rx;
    }
    #pragma unroll
    for (int nt = 0; nt < 16; ++nt) {
        const int n = nbase + nt * 16 + ln;
        #pragma unroll
        for (int v = 0; v < 8; ++v) {
            const int t = t0 + g * 16 + v + 8 * half;
            const float p = pastB[(size_t)t * D_VLM + n];
            outB[(size_t)t * D_VLM + n] = Sf[v] * (Af[v] * p + Bf[v] * acc[nt][v]);
        }
    }
}

extern "C" void kernel_launch(void* const* d_in, const int* in_sizes, int n_in,
                              void* d_out, int out_size, void* d_ws, size_t ws_size,
                              hipStream_t stream) {
    (void)in_sizes; (void)n_in; (void)out_size; (void)d_ws; (void)ws_size;
    const float* past = (const float*)d_in[0];
    const float* obs  = (const float*)d_in[1];
    const int*   cat  = (const int*)d_in[2];
    const float* W    = (const float*)d_in[3];
    const float* bias = (const float*)d_in[4];
    const float* alog = (const float*)d_in[5];
    const float* trms = (const float*)d_in[6];
    float* out = (float*)d_out;

    const size_t smem = 2 * BTSZ + 2 * ALSZ + 128 * 4;   // 300032 B (<320KB WGP LDS)
    (void)hipFuncSetAttribute((const void*)vlm_fused_kernel,
                              hipFuncAttributeMaxDynamicSharedMemorySize, (int)smem);
    vlm_fused_kernel<<<dim3(64 * 16), dim3(512), smem, stream>>>(
        past, obs, cat, W, bias, alog, trms, out);
}
